// ResBlock_12111807775289
// MI455X (gfx1250) — matrix operand order repaired and checked by reading, not compiled, not run on hardware
//
#include <hip/hip_runtime.h>
#include <math.h>

typedef unsigned int   u32;
typedef unsigned short u16;
typedef __bf16 v16bf __attribute__((ext_vector_type(16)));
typedef float  v8f   __attribute__((ext_vector_type(8)));
typedef u32    u32x4 __attribute__((ext_vector_type(4)));

#define B_   8
#define C_   64
#define H_   128
#define W_   128
#define HW_  (H_*W_)
#define HP_  130   // padded
#define NCH  18    // K chunks of 32 (K = 64 ch * 9 taps = 576)

// ---------- helpers ----------
__device__ __forceinline__ u16 f2bf(float f) {
    u32 u = __float_as_uint(f);
    u32 r = ((u >> 16) & 1u) + 0x7FFFu;   // round-to-nearest-even
    u += r;
    return (u16)(u >> 16);
}

// ---------- weight pack: w (O=64,I=64,3,3) f32 -> per-lane B fragments ----------
// wpack[((chunk*4 + nt)*32 + lane)*8 + j] : u32 = bf16 pair {K=Kb+2j+(j>=4?8:0), +1}
// lane<16: N = nt*16+lane, Kb = chunk*32 ; lane>=16: N = nt*16+lane-16, Kb = chunk*32+8
__global__ void pack_w_kernel(const float* __restrict__ w, u32* __restrict__ wpack) {
    int gid = blockIdx.x * 256 + threadIdx.x;
    if (gid >= 4*18*32*8) return;
    int j     = gid & 7;
    int lane  = (gid >> 3) & 31;
    int blk   = gid >> 8;              // 0..71  = chunk*4 + nt
    int nt    = blk & 3;
    int chunk = blk >> 2;
    int n  = (nt << 4) + (lane & 15);
    int Kb = chunk * 32 + ((lane & 16) ? 8 : 0);
    u32 r = 0;
    for (int h = 0; h < 2; ++h) {
        int k   = Kb + 2*j + (j >= 4 ? 8 : 0) + h;
        int tap = k >> 6, ci = k & 63;
        int dy = tap / 3, dx = tap - dy*3;
        float wv = w[(((n*64) + ci)*3 + dy)*3 + dx];
        r |= (u32)f2bf(wv) << (16*h);
    }
    wpack[gid] = r;
}

// ---------- zero-pad + f32->bf16 NCHW -> padded NHWC ----------
__global__ void pad_x_kernel(const float* __restrict__ x, u16* __restrict__ xpad) {
    int b = blockIdx.x / HP_, yp = blockIdx.x % HP_, xp = threadIdx.x;
    if (xp >= HP_) return;
    u16* dst = xpad + (((size_t)b*HP_ + yp)*HP_ + xp)*64;
    if (yp == 0 || yp == HP_-1 || xp == 0 || xp == HP_-1) {
        for (int c = 0; c < 64; ++c) dst[c] = 0;
        return;
    }
    int y = yp - 1, xx = xp - 1;
    for (int c = 0; c < 64; ++c)
        dst[c] = f2bf(x[(((size_t)b*64 + c)*H_ + y)*W_ + xx]);
}

// ---------- WMMA implicit-GEMM 3x3 conv ----------
// One wave owns one 16-pixel M-tile and accumulates all four 16-channel
// N-tiles: 1 A-fragment load feeds 4 WMMAs (72 WMMAs per wave).
// MODE 0: out = gelu(conv + bias)    MODE 1: out = conv + bias + resid
template<int MODE>
__global__ __launch_bounds__(128)
void conv3x3_wmma(const u16* __restrict__ xpad, const u32* __restrict__ wpack,
                  const float* __restrict__ bias, const float* __restrict__ resid,
                  float* __restrict__ out)
{
    const int lane = threadIdx.x & 31;
    const int wv   = threadIdx.x >> 5;            // wave -> M-tile within block
    const int mt   = (blockIdx.x << 2) | wv;      // 0..8191 : 16-pixel row tile
    const int x0 = (mt & 7) << 4;
    const int y  = (mt >> 3) & 127;
    const int b  = mt >> 10;
    const bool hi = lane >= 16;
    const int  m  = lane & 15;

    v8f acc0 = {}, acc1 = {}, acc2 = {}, acc3 = {};
    #pragma unroll
    for (int chunk = 0; chunk < NCH; ++chunk) {
        const int tap = chunk >> 1;
        const int dy = tap / 3, dx = tap - dy*3;
        const int ci0 = (chunk & 1) << 5;
        // A fragment: lanes<16 K {0..7,16..23}, lanes>=16 K {8..15,24..31}
        const u16* abase = xpad +
            ((((size_t)b*HP_ + (y+dy))*HP_ + (x0+m+dx))*64 + ci0 + (hi ? 8 : 0));
        union { u32x4 q[2]; v16bf v; } af;
        af.q[0] = *(const u32x4*)(abase);
        af.q[1] = *(const u32x4*)(abase + 16);
        // B fragments for the 4 N-tiles: contiguous 4 KB region per chunk
        const u32* wchunk = wpack + ((size_t)chunk << 10);   // chunk*4*32*8
        #pragma unroll
        for (int nt = 0; nt < 4; ++nt) {
            union { u32x4 q[2]; v16bf v; } bfr;
            const u32x4* wb = (const u32x4*)(wchunk + (((nt << 5) + lane) << 3));
            bfr.q[0] = wb[0];
            bfr.q[1] = wb[1];
            if      (nt == 0) acc0 = __builtin_amdgcn_wmma_f32_16x16x32_bf16(
                                         false, af.v, false, bfr.v, (short)0, acc0, false, false);
            else if (nt == 1) acc1 = __builtin_amdgcn_wmma_f32_16x16x32_bf16(
                                         false, af.v, false, bfr.v, (short)0, acc1, false, false);
            else if (nt == 2) acc2 = __builtin_amdgcn_wmma_f32_16x16x32_bf16(
                                         false, af.v, false, bfr.v, (short)0, acc2, false, false);
            else              acc3 = __builtin_amdgcn_wmma_f32_16x16x32_bf16(
                                         false, af.v, false, bfr.v, (short)0, acc3, false, false);
        }
    }
    #pragma unroll
    for (int nt = 0; nt < 4; ++nt) {
        const v8f acc = (nt == 0) ? acc0 : (nt == 1) ? acc1 : (nt == 2) ? acc2 : acc3;
        const int n = (nt << 4) + m;              // output channel (D: N = lane&15)
        const float bv = bias[n];
        #pragma unroll
        for (int r = 0; r < 8; ++r) {             // D: M = r + 8*(lane>=16)
            const int xcol = x0 + r + (hi ? 8 : 0);
            const size_t idx = (((size_t)b*64 + n)*H_ + y)*W_ + xcol;
            float v = acc[r] + bv;
            if (MODE == 0) v = 0.5f * v * (1.0f + erff(v * 0.70710678118654752f));
            else           v = v + resid[idx];
            out[idx] = v;
        }
    }
}

// ---------- global average pool over HxW for all 64 channels ----------
__global__ void gap_kernel(const float* __restrict__ a, float* __restrict__ outm) {
    const int bc = blockIdx.x;                     // (b*64 + c)
    const float* p = a + (size_t)bc * HW_;
    float s = 0.f;
    for (int i = threadIdx.x; i < HW_; i += 256) s += p[i];
    __shared__ float sm[256];
    sm[threadIdx.x] = s; __syncthreads();
    for (int st = 128; st > 0; st >>= 1) {
        if (threadIdx.x < st) sm[threadIdx.x] += sm[threadIdx.x + st];
        __syncthreads();
    }
    if (threadIdx.x == 0) outm[bc] = sm[0] * (1.f / HW_);
}

// ---------- per-batch coefficients: dyn softmax weights + SF attention ----------
__global__ __launch_bounds__(32)
void coef_kernel(const float* __restrict__ gapv,
    const float* __restrict__ cw3, const float* __restrict__ g3, const float* __restrict__ bt3,
    const float* __restrict__ m3,  const float* __restrict__ v3,
    const float* __restrict__ fcw3, const float* __restrict__ fcb3,
    const float* __restrict__ hiw3, const float* __restrict__ hib3,
    const float* __restrict__ low3, const float* __restrict__ lob3,
    const float* __restrict__ cw5, const float* __restrict__ g5, const float* __restrict__ bt5,
    const float* __restrict__ m5,  const float* __restrict__ v5,
    const float* __restrict__ fcw5, const float* __restrict__ fcb5,
    const float* __restrict__ hiw5, const float* __restrict__ hib5,
    const float* __restrict__ low5, const float* __restrict__ lob5,
    float* __restrict__ wc3, float* __restrict__ wc5,
    float* __restrict__ ha3, float* __restrict__ la3,
    float* __restrict__ ha5, float* __restrict__ la5)
{
    const int b = blockIdx.x >> 1, br = blockIdx.x & 1, t = threadIdx.x;
    const float* gp  = gapv + b*64 + br*32;
    const float* cw  = br ? cw5  : cw3;  const float* gg  = br ? g5   : g3;
    const float* bb  = br ? bt5  : bt3;  const float* mm  = br ? m5   : m3;
    const float* vv  = br ? v5   : v3;   const float* fcw = br ? fcw5 : fcw3;
    const float* fcb = br ? fcb5 : fcb3; const float* hiw = br ? hiw5 : hiw3;
    const float* hib = br ? hib5 : hib3; const float* lw  = br ? low5 : low3;
    const float* lb  = br ? lob5 : lob3;
    const int k2 = br ? 25 : 9, nK = k2 * 8;

    __shared__ float lf[200], fz[32], att[64], red[2];

    for (int i = t; i < nK; i += 32) {
        float s = 0.f;
        for (int c = 0; c < 32; ++c) s += gp[c] * cw[i*32 + c];
        lf[i] = gg[i] * (s - mm[i]) * rsqrtf(vv[i] + 1e-5f) + bb[i];
    }
    __syncthreads();
    if (t < 8) {   // per-group softmax over kernel positions
        float mx = -3.4e38f;
        for (int k = 0; k < k2; ++k) mx = fmaxf(mx, lf[t*k2 + k]);
        float e[25], sum = 0.f;
        for (int k = 0; k < k2; ++k) { e[k] = expf(lf[t*k2 + k] - mx); sum += e[k]; }
        float inv = 1.f / sum;
        float* wo = (br ? wc5 + b*200 : wc3 + b*72) + t*k2;
        for (int k = 0; k < k2; ++k) wo[k] = e[k] * inv;
    }
    {   // fz = z @ fc_w.T + fc_b     (z == branch GAP since low+high == x)
        float s = 0.f;
        for (int c = 0; c < 32; ++c) s += gp[c] * fcw[t*32 + c];
        fz[t] = s + fcb[t];
    }
    __syncthreads();
    {
        float sh = 0.f, sl = 0.f;
        for (int j = 0; j < 32; ++j) { float f = fz[j]; sh += f*hiw[t*32+j]; sl += f*lw[t*32+j]; }
        att[t]      = sh + hib[t];
        att[32 + t] = sl + lb[t];
    }
    __syncthreads();
    if (t == 0) { float mx = -3.4e38f; for (int i = 0; i < 64; ++i) mx = fmaxf(mx, att[i]); red[0] = mx; }
    __syncthreads();
    float e1 = expf(att[t] - red[0]), e2 = expf(att[32 + t] - red[0]);
    att[t] = e1; att[32 + t] = e2;
    __syncthreads();
    if (t == 0) { float s = 0.f; for (int i = 0; i < 64; ++i) s += att[i]; red[1] = 1.f / s; }
    __syncthreads();
    (br ? ha5 : ha3)[b*32 + t] = e1 * red[1];
    (br ? la5 : la3)[b*32 + t] = e2 * red[1];
}

// ---------- dynamic-filter fused output: fused = x*ha + low*(la-ha) ----------
__global__ __launch_bounds__(128)
void dyn_fused_kernel(const float* __restrict__ act,
                      const float* __restrict__ wc3, const float* __restrict__ wc5,
                      const float* __restrict__ ha3, const float* __restrict__ la3,
                      const float* __restrict__ ha5, const float* __restrict__ la5,
                      float* __restrict__ fused)
{
    const int r  = blockIdx.x;                 // b*64*128 + cg*128 + y
    const int y  = r & 127;
    const int cg = (r >> 7) & 63;
    const int b  = r >> 13;
    const int x  = threadIdx.x;
    const int br = cg >> 5, cl = cg & 31, g = cl >> 2;
    const int k  = br ? 5 : 3, p = br ? 2 : 1, k2 = k*k;
    const float* wcf = br ? (wc5 + b*200 + g*25) : (wc3 + b*72 + g*9);
    const float hav = (br ? ha5 : ha3)[b*32 + cl];
    const float lav = (br ? la5 : la3)[b*32 + cl];
    const float* ap = act + ((size_t)b*64 + cg) * HW_;
    float low = 0.f;
    for (int tpos = 0; tpos < k2; ++tpos) {
        int dy = tpos / k, dx = tpos - dy*k;
        int qy = y + dy - p; qy = qy < 0 ? -qy : (qy > 127 ? 254 - qy : qy);  // reflect
        int qx = x + dx - p; qx = qx < 0 ? -qx : (qx > 127 ? 254 - qx : qx);
        low += wcf[tpos] * ap[qy*W_ + qx];
    }
    const float a0 = ap[y*W_ + x];
    fused[((size_t)b*64 + cg)*HW_ + y*W_ + x] = a0*hav + low*(lav - hav);
}

// ---------- 1x1 conv (32->32) per branch ----------
__global__ __launch_bounds__(128)
void pw_conv_kernel(const float* __restrict__ fused,
                    const float* __restrict__ ow3, const float* __restrict__ ob3,
                    const float* __restrict__ ow5, const float* __restrict__ ob5,
                    float* __restrict__ y1)
{
    const int r = blockIdx.x;                 // b*2*128 + br*128 + y
    const int y = r & 127, br = (r >> 7) & 1, b = r >> 8;
    __shared__ float ws[1024], wb[32];
    const float* ow = br ? ow5 : ow3;
    const float* ob = br ? ob5 : ob3;
    for (int i = threadIdx.x; i < 1024; i += 128) ws[i] = ow[i];
    if (threadIdx.x < 32) wb[threadIdx.x] = ob[threadIdx.x];
    __syncthreads();
    const int x = threadIdx.x;
    float in[32];
    const float* fp = fused + (((size_t)b*64 + br*32)*H_ + y)*W_ + x;
    for (int c = 0; c < 32; ++c) in[c] = fp[(size_t)c * HW_];
    float* op = y1 + (((size_t)b*64 + br*32)*H_ + y)*W_ + x;
    for (int o = 0; o < 32; ++o) {
        float s = wb[o];
        const float* wr = ws + o*32;
        for (int c = 0; c < 32; ++c) s += wr[c] * in[c];
        op[(size_t)o * HW_] = s;
    }
}

// ---------- full-image means (ch 0..31) + quadrant means (ch 32..63) ----------
__global__ void post_means_kernel(const float* __restrict__ y1,
                                  float* __restrict__ gm, float* __restrict__ qm)
{
    const int blk = blockIdx.x;
    __shared__ float sm[256];
    float s = 0.f; float* dst; float scale;
    if (blk < 256) {                              // gm[b][c], c<32
        const int b = blk >> 5, c = blk & 31;
        const float* p = y1 + ((size_t)b*64 + c) * HW_;
        for (int i = threadIdx.x; i < HW_; i += 256) s += p[i];
        dst = gm + blk; scale = 1.f / HW_;
    } else {                                      // qm[b][cl][q]
        const int i = blk - 256;
        const int b = i >> 7, rem = i & 127, cl = rem >> 2, q = rem & 3;
        const int p1 = q >> 1, p2 = q & 1;
        const float* p = y1 + ((size_t)b*64 + 32 + cl)*HW_ + p1*64*W_ + p2*64;
        for (int e = threadIdx.x; e < 4096; e += 256) s += p[(e >> 6)*W_ + (e & 63)];
        dst = qm + i; scale = 1.f / 4096.f;
    }
    sm[threadIdx.x] = s; __syncthreads();
    for (int st = 128; st > 0; st >>= 1) {
        if (threadIdx.x < st) sm[threadIdx.x] += sm[threadIdx.x + st];
        __syncthreads();
    }
    if (threadIdx.x == 0) *dst = sm[0] * scale;
}

// ---------- gap/patch_ap affine + write padded NHWC bf16 for conv2 ----------
__global__ void transform_pad_kernel(const float* __restrict__ y1,
                                     const float* __restrict__ gm, const float* __restrict__ qm,
                                     const float* __restrict__ gap_d, const float* __restrict__ gap_h,
                                     const float* __restrict__ pa_h, const float* __restrict__ pa_l,
                                     u16* __restrict__ xpad2)
{
    const int b = blockIdx.x / HP_, yp = blockIdx.x % HP_, xp = threadIdx.x;
    if (xp >= HP_) return;
    u16* dst = xpad2 + (((size_t)b*HP_ + yp)*HP_ + xp)*64;
    if (yp == 0 || yp == HP_-1 || xp == 0 || xp == HP_-1) {
        for (int c = 0; c < 64; ++c) dst[c] = 0;
        return;
    }
    const int y = yp - 1, x = xp - 1;
    const int p1 = y >> 6, p2 = x >> 6, q = (p1 << 1) + p2;
    for (int c = 0; c < 64; ++c) {
        const float v = y1[(((size_t)b*64 + c)*H_ + y)*W_ + x];
        float o;
        if (c < 32) {                                       // _gap
            const float xd = gm[b*32 + c];
            o = (v - xd) * (gap_h[c] + 1.f) + xd * gap_d[c];
        } else {                                            // _patch_ap (quadrants)
            const int cl = c - 32;
            const float lwv = qm[((b*32 + cl) << 2) + q];
            const int idx = (cl << 2) + q;                  // (c p1 p2) ordering
            o = (v - lwv) * pa_h[idx] + lwv * pa_l[idx];
        }
        dst[c] = f2bf(o);
    }
}

// =============================== launcher ===============================
extern "C" void kernel_launch(void* const* d_in, const int* in_sizes, int n_in,
                              void* d_out, int out_size, void* d_ws, size_t ws_size,
                              hipStream_t stream)
{
    const float* x    = (const float*)d_in[0];
    const float* w1   = (const float*)d_in[1];
    const float* b1   = (const float*)d_in[2];
    const float* w2   = (const float*)d_in[3];
    const float* b2   = (const float*)d_in[4];
    // d3 params: 5..17 ; d5 params: 18..30
    const float* d3p[13]; for (int i = 0; i < 13; ++i) d3p[i] = (const float*)d_in[5 + i];
    const float* d5p[13]; for (int i = 0; i < 13; ++i) d5p[i] = (const float*)d_in[18 + i];
    const float* gap_d = (const float*)d_in[31];
    const float* gap_h = (const float*)d_in[32];
    const float* pa_h  = (const float*)d_in[33];
    const float* pa_l  = (const float*)d_in[34];
    float* out = (float*)d_out;

    // workspace layout (xpad reused for conv2 input, act1 reused as y1)
    char* ws = (char*)d_ws;
    const size_t SZ_XPAD = (size_t)B_*HP_*HP_*64*2;        // 17,305,600
    const size_t SZ_ACT  = (size_t)B_*64*HW_*4;            // 33,554,432
    size_t off = 0;
    u16*   xpad  = (u16*)(ws + off); off += (SZ_XPAD + 255) & ~255ull;
    float* act1  = (float*)(ws + off); off += SZ_ACT;      // also y1
    float* fused = (float*)(ws + off); off += SZ_ACT;
    u32*   wpk1  = (u32*)(ws + off); off += 18432*4;
    u32*   wpk2  = (u32*)(ws + off); off += 18432*4;
    float* gapv  = (float*)(ws + off); off += 512*4;
    float* wc3   = (float*)(ws + off); off += 8*72*4;
    float* wc5   = (float*)(ws + off); off += 8*200*4;
    float* ha3   = (float*)(ws + off); off += 256*4;
    float* la3   = (float*)(ws + off); off += 256*4;
    float* ha5   = (float*)(ws + off); off += 256*4;
    float* la5   = (float*)(ws + off); off += 256*4;
    float* gm    = (float*)(ws + off); off += 256*4;
    float* qm    = (float*)(ws + off); off += 1024*4;
    float* y1    = act1;
    u16*   xpad2 = xpad;

    pack_w_kernel<<<72, 256, 0, stream>>>(w1, wpk1);
    pack_w_kernel<<<72, 256, 0, stream>>>(w2, wpk2);
    pad_x_kernel<<<B_*HP_, 160, 0, stream>>>(x, xpad);
    conv3x3_wmma<0><<<2048, 128, 0, stream>>>(xpad, wpk1, b1, nullptr, act1);
    gap_kernel<<<512, 256, 0, stream>>>(act1, gapv);
    coef_kernel<<<16, 32, 0, stream>>>(gapv,
        d3p[0], d3p[1], d3p[2], d3p[3], d3p[4], d3p[5], d3p[6], d3p[7], d3p[8], d3p[9], d3p[10],
        d5p[0], d5p[1], d5p[2], d5p[3], d5p[4], d5p[5], d5p[6], d5p[7], d5p[8], d5p[9], d5p[10],
        wc3, wc5, ha3, la3, ha5, la5);
    dyn_fused_kernel<<<B_*64*H_, 128, 0, stream>>>(act1, wc3, wc5, ha3, la3, ha5, la5, fused);
    pw_conv_kernel<<<B_*2*H_, 128, 0, stream>>>(fused, d3p[11], d3p[12], d5p[11], d5p[12], y1);
    post_means_kernel<<<1280, 256, 0, stream>>>(y1, gm, qm);
    transform_pad_kernel<<<B_*HP_, 160, 0, stream>>>(y1, gm, qm, gap_d, gap_h, pa_h, pa_l, xpad2);
    conv3x3_wmma<1><<<2048, 128, 0, stream>>>(xpad2, wpk2, b2, x, out);
}
